// GPT2LMHeadModel_62818191671339
// MI455X (gfx1250) — compile-verified, matrix operand order checked by reading, not run on hardware
//
#include <hip/hip_runtime.h>
#include <hip/hip_bf16.h>

typedef __bf16 bf16;
typedef __attribute__((ext_vector_type(16))) __bf16 v16bf;
typedef __attribute__((ext_vector_type(8)))  float  v8f;

#define DMODEL 768
#define NHEAD  12
#define HDIM   64
#define SEQ    1024
#define BATCH  2
#define NLAYER 12
#define ROWS   (BATCH*SEQ)   // 2048
#define VOCAB  50257

// ---- WMMA helpers -----------------------------------------------------------
// A-fragment K mapping for 16-bit 16x32 A (ISA 7.12.2):
// lanes 0-15: VGPR0..3 -> K=0..7, VGPR4..7 -> K=16..23
// lanes 16-31: VGPR0..3 -> K=8..15, VGPR4..7 -> K=24..31
__device__ __forceinline__ int kmap(int j, int lane) {
  int hi8 = (lane & 16) ? 8 : 0;
  return (j < 4) ? (hi8 + 2 * j) : (16 + hi8 + 2 * (j - 4));
}

union AFrag { v16bf v; unsigned u[8]; };
union Pk4 { bf16 h[4]; uint2 u2; };

__device__ __forceinline__ v8f wmma_bf16(v16bf a, v16bf b, v8f c) {
  return __builtin_amdgcn_wmma_f32_16x16x32_bf16(
      false, a, false, b, (short)0, c, false, false);
}

// ---- Embedding: x = wte[id] + wpe[s] ---------------------------------------
__global__ void embed_kernel(const int* __restrict__ ids,
                             const float* __restrict__ wte,
                             const float* __restrict__ wpe,
                             float* __restrict__ x) {
  int row = blockIdx.x;
  int s   = row % SEQ;
  int id  = ids[row];
  const float* we = wte + (size_t)id * DMODEL;
  const float* pe = wpe + (size_t)s * DMODEL;
  float* xr = x + (size_t)row * DMODEL;
  for (int d = threadIdx.x; d < DMODEL; d += blockDim.x)
    xr[d] = we[d] + pe[d];
}

// ---- LayerNorm (wave32 per row) -> bf16 out --------------------------------
__global__ __launch_bounds__(256) void ln_kernel(const float* __restrict__ x,
                                                 const float* __restrict__ g,
                                                 const float* __restrict__ b,
                                                 bf16* __restrict__ out) {
  int wave = threadIdx.x >> 5, lane = threadIdx.x & 31;
  int row = blockIdx.x * 8 + wave;
  const float* xr = x + (size_t)row * DMODEL;
  float v[24];
  float s = 0.f;
#pragma unroll
  for (int j = 0; j < 24; j++) { v[j] = xr[lane + 32 * j]; s += v[j]; }
#pragma unroll
  for (int m = 1; m < 32; m <<= 1) s += __shfl_xor(s, m, 32);
  float mean = s * (1.0f / 768.0f);
  float q = 0.f;
#pragma unroll
  for (int j = 0; j < 24; j++) { float d = v[j] - mean; q += d * d; }
#pragma unroll
  for (int m = 1; m < 32; m <<= 1) q += __shfl_xor(q, m, 32);
  float rstd = rsqrtf(q * (1.0f / 768.0f) + 1e-5f);
  bf16* orow = out + (size_t)row * DMODEL;
#pragma unroll
  for (int j = 0; j < 24; j++) {
    int idx = lane + 32 * j;
    orow[idx] = (bf16)((v[j] - mean) * rstd * g[idx] + b[idx]);
  }
}

// ---- Generic tiled GEMM: C = act(A(bf16)[M,K] @ B(f32) + bias) (+res) ------
// BT: B given as [N,K] row-major (B^T, used for tied LM head with wte[V,D])
// otherwise B is [K,N] (N then always a multiple of 128 -> no guards).
// Tile 128x128x32, 8 waves, each wave 64x32 (8 WMMA / K-step).
// LDS double-buffered (ping-pong): one barrier per K-step, global staging of
// tile k+1 overlapped with the WMMAs of tile k. 128-bit global loads.
template <bool BT, bool BIAS, bool GELU, bool RES>
__global__ __launch_bounds__(256) void gemm_kernel(
    const bf16* __restrict__ A, const float* __restrict__ B,
    const float* __restrict__ bias, const float* __restrict__ res,
    float* __restrict__ outF, bf16* __restrict__ outB, int M, int N, int K) {
  __shared__ bf16 As[2][128][40];   // [M][K], pad 40: 16B-aligned vector stores
  __shared__ bf16 Bs[2][128][40];   // stored as B^T tile: [N][K]
  int t = threadIdx.x;
  int lane = t & 31, wave = t >> 5;
  int wm = wave >> 2, wn = wave & 3;
  int ln16 = lane & 15;
  int rhi = (lane & 16) ? 8 : 0;
  int m0 = blockIdx.y * 128, n0 = blockIdx.x * 128;

  // staging coordinates (fixed per thread)
  int ar = t >> 1;            // A row 0..127
  int ac = (t & 1) << 4;      // A col half: 0 or 16 (16 bf16 = 32B per thread)

  auto stage = [&](int buf, int k0) {
    // ---- A: 128x32 bf16, 2x uint4 (32B) per thread, coalesced ----
    const uint4* ag = (const uint4*)(A + (size_t)(m0 + ar) * K + k0 + ac);
    uint4 a0 = ag[0];
    uint4 a1 = ag[1];
    if (k0 + 32 < K)  // prefetch next A tile (-> global_prefetch_b8)
      __builtin_prefetch(A + (size_t)(m0 + ar) * K + k0 + 32 + ac, 0, 1);
    *(uint4*)&As[buf][ar][ac]     = a0;
    *(uint4*)&As[buf][ar][ac + 8] = a1;
    // ---- B: cvt f32->bf16 into [N][K], float4 global loads ----
    if (BT) {
#pragma unroll
      for (int j = 0; j < 4; j++) {
        int i = t + 256 * j;
        int n = i >> 3;            // 0..127
        int c = (i & 7) << 2;      // 0,4,...,28
        float4 f = make_float4(0.f, 0.f, 0.f, 0.f);
        if (n0 + n < N)
          f = *(const float4*)(B + (size_t)(n0 + n) * K + k0 + c);
        Pk4 pk;
        pk.h[0] = (bf16)f.x; pk.h[1] = (bf16)f.y;
        pk.h[2] = (bf16)f.z; pk.h[3] = (bf16)f.w;
        *(uint2*)&Bs[buf][n][c] = pk.u2;
      }
    } else {
#pragma unroll
      for (int j = 0; j < 4; j++) {
        int i = t + 256 * j;
        int n4 = (i & 31) << 2;    // 0,4,...,124
        int c = i >> 5;            // 0..31
        float4 f = *(const float4*)(B + (size_t)(k0 + c) * N + n0 + n4);
        Bs[buf][n4][c]     = (bf16)f.x;
        Bs[buf][n4 + 1][c] = (bf16)f.y;
        Bs[buf][n4 + 2][c] = (bf16)f.z;
        Bs[buf][n4 + 3][c] = (bf16)f.w;
      }
    }
  };

  v8f zero8 = {0, 0, 0, 0, 0, 0, 0, 0};
  v8f acc[4][2];
#pragma unroll
  for (int i = 0; i < 4; i++)
#pragma unroll
    for (int j = 0; j < 2; j++) acc[i][j] = zero8;

  int nk = K >> 5;
  stage(0, 0);
  for (int kt = 0; kt < nk; kt++) {
    int buf = kt & 1;
    __syncthreads();                       // buf ready; other buf consumed
    if (kt + 1 < nk) stage(buf ^ 1, (kt + 1) << 5);

    AFrag af[4], bfr[2];
#pragma unroll
    for (int i = 0; i < 4; i++) {
      const bf16* p = &As[buf][wm * 64 + i * 16 + ln16][0];
#pragma unroll
      for (int j = 0; j < 8; j++)
        af[i].u[j] = *(const unsigned*)(p + kmap(j, lane));
    }
#pragma unroll
    for (int jn = 0; jn < 2; jn++) {
      const bf16* p = &Bs[buf][wn * 32 + jn * 16 + ln16][0];
#pragma unroll
      for (int j = 0; j < 8; j++)
        bfr[jn].u[j] = *(const unsigned*)(p + kmap(j, lane));
    }
#pragma unroll
    for (int i = 0; i < 4; i++)
#pragma unroll
      for (int jn = 0; jn < 2; jn++)
        acc[i][jn] = wmma_bf16(af[i].v, bfr[jn].v, acc[i][jn]);
  }

#pragma unroll
  for (int i = 0; i < 4; i++) {
#pragma unroll
    for (int jn = 0; jn < 2; jn++) {
      int col = n0 + wn * 32 + jn * 16 + ln16;
      if (col < N) {
#pragma unroll
        for (int r = 0; r < 8; r++) {
          int row = m0 + wm * 64 + i * 16 + r + rhi;
          float vv = acc[i][jn][r];
          if (BIAS) vv += bias[col];
          if (GELU) vv = 0.5f * vv * (1.0f + erff(vv * 0.70710678118f));
          if (RES)  vv += res[(size_t)row * N + col];
          if (outF) outF[(size_t)row * N + col] = vv;
          if (outB) outB[(size_t)row * N + col] = (bf16)vv;
        }
      }
    }
  }
}

// ---- Fused flash attention (causal), WMMA for QK^T and PV ------------------
// Block: 128 queries (8 waves x 16 rows) for one (b,h). K/V staged per 128-key
// superblock in LDS; online softmax; P transposed C->A layout via LDS scratch.
__global__ __launch_bounds__(256) void attn_kernel(const bf16* __restrict__ qkv,
                                                   bf16* __restrict__ out) {
  __shared__ bf16 Ks[128][68];      // [key][hd], B^T layout for scores
  __shared__ bf16 Vt[64][132];      // [hd][key], B^T layout for P@V
  __shared__ bf16 Ps[8][16][34];    // per-wave P tile [qrow][key 0..31]
  int t = threadIdx.x;
  int lane = t & 31, wave = t >> 5;
  int ln16 = lane & 15;
  int rhi = (lane & 16) ? 8 : 0;
  int bh = blockIdx.y;
  int h = bh % NHEAD, b = bh / NHEAD;
  int sbase = blockIdx.x * 128;
  int kmax = sbase + 128;
  const unsigned* qkv32 = (const unsigned*)qkv;

  // Q fragments (2 k-steps over HD=64), stay in registers for whole kernel
  int sq = sbase + wave * 16 + ln16;
  AFrag qf[2];
#pragma unroll
  for (int ks = 0; ks < 2; ks++)
#pragma unroll
    for (int j = 0; j < 8; j++) {
      int hd = kmap(j, lane) + 32 * ks;
      qf[ks].u[j] = qkv32[((size_t)(b * SEQ + sq) * 2304 + h * 64 + hd) >> 1];
    }

  v8f zero8 = {0, 0, 0, 0, 0, 0, 0, 0};
  v8f oacc[4];
#pragma unroll
  for (int j = 0; j < 4; j++) oacc[j] = zero8;
  float m_[8], l_[8];
#pragma unroll
  for (int r = 0; r < 8; r++) { m_[r] = -INFINITY; l_[r] = 0.f; }

  for (int kb0 = 0; kb0 < kmax; kb0 += 128) {
    __syncthreads();
#pragma unroll
    for (int j = 0; j < 16; j++) {    // stage K [key][hd] and V^T [hd][key]
      int i = t + 256 * j;
      int key = i >> 5;
      int hd2 = (i & 31) << 1;
      size_t base = (size_t)(b * SEQ + kb0 + key) * 2304 + h * 64;
      *(unsigned*)&Ks[key][hd2] = qkv32[(base + 768 + hd2) >> 1];
      unsigned uv = qkv32[(base + 1536 + hd2) >> 1];
      Vt[hd2][key]     = ((const bf16*)&uv)[0];
      Vt[hd2 + 1][key] = ((const bf16*)&uv)[1];
    }
    __syncthreads();

    int wq_last = sbase + wave * 16 + 15;
    for (int sub = 0; sub < 4; sub++) {
      int kb = kb0 + sub * 32;
      if (kb > wq_last) break;        // fully masked for this wave

      // scores: 2 n-tiles x 2 k-steps
      v8f sacc[2];
#pragma unroll
      for (int nt = 0; nt < 2; nt++) {
        sacc[nt] = zero8;
#pragma unroll
        for (int ks = 0; ks < 2; ks++) {
          AFrag bF;
          const bf16* p = &Ks[sub * 32 + nt * 16 + ln16][32 * ks];
#pragma unroll
          for (int j = 0; j < 8; j++)
            bF.u[j] = *(const unsigned*)(p + kmap(j, lane));
          sacc[nt] = wmma_bf16(qf[ks].v, bF.v, sacc[nt]);
        }
      }

      // scale + causal mask + online softmax update
      float pbuf[2][8];
#pragma unroll
      for (int r = 0; r < 8; r++) {
        int q_s = sbase + wave * 16 + r + rhi;
        float s0 = sacc[0][r] * 0.125f;
        float s1 = sacc[1][r] * 0.125f;
        if (kb + ln16 > q_s)      s0 = -INFINITY;
        if (kb + 16 + ln16 > q_s) s1 = -INFINITY;
        float mx = fmaxf(s0, s1);
#pragma unroll
        for (int mk = 1; mk < 16; mk <<= 1) mx = fmaxf(mx, __shfl_xor(mx, mk, 32));
        float mnew = fmaxf(m_[r], mx);
        float corr = __expf(m_[r] - mnew);
        m_[r] = mnew;
        float p0 = __expf(s0 - mnew);
        float p1 = __expf(s1 - mnew);
        float rs = p0 + p1;
#pragma unroll
        for (int mk = 1; mk < 16; mk <<= 1) rs += __shfl_xor(rs, mk, 32);
        l_[r] = l_[r] * corr + rs;
#pragma unroll
        for (int j = 0; j < 4; j++) oacc[j][r] *= corr;
        pbuf[0][r] = p0; pbuf[1][r] = p1;
      }

      // P: C-layout -> A-layout via wave-private LDS scratch
#pragma unroll
      for (int r = 0; r < 8; r++) {
        Ps[wave][r + rhi][ln16]      = (bf16)pbuf[0][r];
        Ps[wave][r + rhi][16 + ln16] = (bf16)pbuf[1][r];
      }
      AFrag pf;
      const bf16* pp = &Ps[wave][ln16][0];
#pragma unroll
      for (int j = 0; j < 8; j++)
        pf.u[j] = *(const unsigned*)(pp + kmap(j, lane));

      // O += P @ V : 4 hd-tiles
#pragma unroll
      for (int j = 0; j < 4; j++) {
        AFrag vF;
        const bf16* vp = &Vt[j * 16 + ln16][sub * 32];
#pragma unroll
        for (int jj = 0; jj < 8; jj++)
          vF.u[jj] = *(const unsigned*)(vp + kmap(jj, lane));
        oacc[j] = wmma_bf16(pf.v, vF.v, oacc[j]);
      }
    }
  }

  // normalize and write to [B,S,D] bf16 (input of projection GEMM)
#pragma unroll
  for (int j = 0; j < 4; j++)
#pragma unroll
    for (int r = 0; r < 8; r++) {
      int s2 = sbase + wave * 16 + r + rhi;
      float val = oacc[j][r] / l_[r];
      out[(size_t)(b * SEQ + s2) * DMODEL + h * 64 + j * 16 + ln16] = (bf16)val;
    }
}

// ---- Host orchestration -----------------------------------------------------
extern "C" void kernel_launch(void* const* d_in, const int* in_sizes, int n_in,
                              void* d_out, int out_size, void* d_ws,
                              size_t ws_size, hipStream_t stream) {
  (void)in_sizes; (void)n_in; (void)out_size; (void)ws_size;
  const int*   ids   = (const int*)d_in[0];
  const float* wte   = (const float*)d_in[1];
  const float* wpe   = (const float*)d_in[2];
  const float* ln1_g = (const float*)d_in[3];
  const float* ln1_b = (const float*)d_in[4];
  const float* Wqkv  = (const float*)d_in[5];
  const float* bqkv  = (const float*)d_in[6];
  const float* Wo    = (const float*)d_in[7];
  const float* bo    = (const float*)d_in[8];
  const float* ln2_g = (const float*)d_in[9];
  const float* ln2_b = (const float*)d_in[10];
  const float* Wfc   = (const float*)d_in[11];
  const float* bfc   = (const float*)d_in[12];
  const float* Wp    = (const float*)d_in[13];
  const float* bp    = (const float*)d_in[14];
  const float* lnf_g = (const float*)d_in[15];
  const float* lnf_b = (const float*)d_in[16];

  char* ws = (char*)d_ws;
  size_t off = 0;
  auto alloc = [&](size_t bytes) {
    void* p = ws + off;
    off += (bytes + 255) & ~((size_t)255);
    return p;
  };
  float* x    = (float*)alloc((size_t)ROWS * DMODEL * 4);       // residual f32
  bf16* hbuf  = (bf16*)alloc((size_t)ROWS * DMODEL * 2);        // LN out
  bf16* qkvb  = (bf16*)alloc((size_t)ROWS * 3 * DMODEL * 2);    // QKV
  bf16* attnb = (bf16*)alloc((size_t)ROWS * DMODEL * 2);        // attn out
  bf16* mlpb  = (bf16*)alloc((size_t)ROWS * 4 * DMODEL * 2);    // MLP hidden

  embed_kernel<<<ROWS, 256, 0, stream>>>(ids, wte, wpe, x);

  for (int l = 0; l < NLAYER; l++) {
    ln_kernel<<<ROWS / 8, 256, 0, stream>>>(x, ln1_g + l * DMODEL,
                                            ln1_b + l * DMODEL, hbuf);
    gemm_kernel<false, true, false, false><<<dim3(18, 16), 256, 0, stream>>>(
        hbuf, Wqkv + (size_t)l * DMODEL * 3 * DMODEL, bqkv + l * 3 * DMODEL,
        nullptr, nullptr, qkvb, ROWS, 3 * DMODEL, DMODEL);
    attn_kernel<<<dim3(SEQ / 128, BATCH * NHEAD), 256, 0, stream>>>(qkvb, attnb);
    gemm_kernel<false, true, false, true><<<dim3(6, 16), 256, 0, stream>>>(
        attnb, Wo + (size_t)l * DMODEL * DMODEL, bo + l * DMODEL, x, x, nullptr,
        ROWS, DMODEL, DMODEL);
    ln_kernel<<<ROWS / 8, 256, 0, stream>>>(x, ln2_g + l * DMODEL,
                                            ln2_b + l * DMODEL, hbuf);
    gemm_kernel<false, true, true, false><<<dim3(24, 16), 256, 0, stream>>>(
        hbuf, Wfc + (size_t)l * DMODEL * 4 * DMODEL, bfc + l * 4 * DMODEL,
        nullptr, nullptr, mlpb, ROWS, 4 * DMODEL, DMODEL);
    gemm_kernel<false, true, false, true><<<dim3(6, 16), 256, 0, stream>>>(
        mlpb, Wp + (size_t)l * 4 * DMODEL * DMODEL, bp + l * DMODEL, x, x,
        nullptr, ROWS, DMODEL, 4 * DMODEL);
  }

  ln_kernel<<<ROWS / 8, 256, 0, stream>>>(x, lnf_g, lnf_b, hbuf);
  // tied LM head: logits = h @ wte^T ; wte is [V, D] which is exactly B^T
  gemm_kernel<true, false, false, false>
      <<<dim3((VOCAB + 127) / 128, 16), 256, 0, stream>>>(
          hbuf, wte, nullptr, nullptr, (float*)d_out, nullptr, ROWS, VOCAB,
          DMODEL);
}
